// FACTORMODEL_56908316672534
// MI455X (gfx1250) — compile-verified, matrix operand order checked by reading, not run on hardware
//
#include <hip/hip_runtime.h>
#include <hip/hip_bf16.h>
#include <stdint.h>
#include <stddef.h>

#define N_ 8192
#define C_ 512
#define H_ 128

typedef __attribute__((ext_vector_type(16))) __bf16 v16bf;
typedef __attribute__((ext_vector_type(8)))  __bf16 v8bf;
typedef __attribute__((ext_vector_type(8)))  float  v8f;

// ---------- helpers ----------
__device__ __forceinline__ __bf16 f2bf(float x) {
  union { float f; uint32_t u; } v; v.f = x;
  uint32_t r = v.u + 0x7FFFu + ((v.u >> 16) & 1u);
  uint16_t h = (uint16_t)(r >> 16);
  return __builtin_bit_cast(__bf16, h);
}

// A-fragment (16x32 bf16) / B-fragment (32x16 via B^T row-major) loader.
// lane l: row = r0 + (l&15), k = k0 + ((l>>4)<<4), 16 contiguous bf16.
__device__ __forceinline__ v16bf ldfrag(const __bf16* __restrict__ M, int ld, int r0, int k0) {
  int lane = threadIdx.x & 31;
  const __bf16* p = M + (size_t)(r0 + (lane & 15)) * ld + (k0 + ((lane >> 4) << 4));
  return *(const v16bf*)p;
}

__device__ __forceinline__ v8f wmma_bf16(v16bf a, v16bf b, v8f c) {
  return __builtin_amdgcn_wmma_f32_16x16x32_bf16(false, a, false, b, (short)0, c, false, false);
}

__device__ __forceinline__ v8f vzero8() {
  v8f z;
#pragma unroll
  for (int r = 0; r < 8; ++r) z[r] = 0.0f;
  return z;
}

// ---------- elementwise / utility kernels ----------
__global__ void __launch_bounds__(256) k_zero(float* p, int n) {
  int i = blockIdx.x * 256 + threadIdx.x;
  if (i < n) p[i] = 0.0f;
}

__global__ void __launch_bounds__(256) k_f2b(const float* __restrict__ in, __bf16* __restrict__ out, int n) {
  int i = blockIdx.x * 256 + threadIdx.x;
  if (i < n) out[i] = f2bf(in[i]);
}

// transpose [R,Cc] f32 -> [Cc,R] bf16
__global__ void __launch_bounds__(256) k_tr_f2b(const float* __restrict__ in, __bf16* __restrict__ out,
                                                int R, int Cc) {
  int idx = blockIdx.x * 256 + threadIdx.x;
  if (idx >= R * Cc) return;
  int r = idx / Cc, c = idx - r * Cc;
  out[(size_t)c * R + r] = f2bf(in[(size_t)r * Cc + c]);
}

__global__ void __launch_bounds__(256) k_sub2(const float* a, const float* b, float* o, int n) {
  int i = blockIdx.x * 256 + threadIdx.x;
  if (i < n) o[i] = a[i] - b[i];
}

__global__ void __launch_bounds__(256) k_indiv(const float* a, const float* b, const float* c,
                                               __bf16* o, int n) {
  int i = blockIdx.x * 256 + threadIdx.x;
  if (i < n) o[i] = f2bf(a[i] - b[i] - c[i]);
}

// ---------- phase A: market-value aggregation ----------
__global__ void __launch_bounds__(256) k_colsum(const int* __restrict__ cm, const float* __restrict__ mv,
                                                float* __restrict__ colsum) {
  int t = threadIdx.x, b = blockIdx.x;   // 256 blocks x 256 thr, 32 rows/block
  float s0 = 0.f, s1 = 0.f;
#pragma unroll 4
  for (int r = 0; r < 32; ++r) {
    int i = b * 32 + r;
    float m = mv[i];
    s0 += (float)cm[(size_t)i * C_ + t]        * m;
    s1 += (float)cm[(size_t)i * C_ + t + 256]  * m;
  }
  atomicAdd(&colsum[t], s0);
  atomicAdd(&colsum[t + 256], s1);
}

// s2c^T [C,N] bf16: cm?mv/(colsum+1):0
__global__ void __launch_bounds__(256) k_s2cT(const int* __restrict__ cm, const float* __restrict__ mv,
                                              const float* __restrict__ colsum, __bf16* __restrict__ out) {
  int idx = blockIdx.x * 256 + threadIdx.x;
  if (idx >= C_ * N_) return;
  int c = idx / N_, i = idx - c * N_;
  float v = cm[(size_t)i * C_ + c] ? (mv[i] / (colsum[c] + 1.0f)) : 0.0f;
  out[idx] = f2bf(v);
}

// ---------- generic WMMA GEMM, 32 rows/wave, double-buffered fragments ----------
__global__ void __launch_bounds__(64) k_gemm(const __bf16* __restrict__ A, const __bf16* __restrict__ Bt,
                                             const float* __restrict__ bias, float* __restrict__ Cf,
                                             __bf16* __restrict__ Cb, int M, int Nc, int K, int act) {
  const int lane = threadIdx.x & 31;
  const int wave = threadIdx.x >> 5;
  const int row0 = blockIdx.x * 64 + wave * 32;
  const int col0 = blockIdx.y * 128;
  v8f acc0[8], acc1[8];
#pragma unroll
  for (int t = 0; t < 8; ++t) { acc0[t] = vzero8(); acc1[t] = vzero8(); }

  v16bf a0 = ldfrag(A, K, row0, 0), a1 = ldfrag(A, K, row0 + 16, 0);
  v16bf b[8];
#pragma unroll
  for (int t = 0; t < 8; ++t) b[t] = ldfrag(Bt, K, col0 + t * 16, 0);

  for (int kc = 0; kc < K; kc += 32) {
    int kn = (kc + 32 < K) ? kc + 32 : kc;          // clamp: branchless last iter
    v16bf na0 = ldfrag(A, K, row0, kn);
    v16bf na1 = ldfrag(A, K, row0 + 16, kn);
    v16bf nb[8];
#pragma unroll
    for (int t = 0; t < 8; ++t) nb[t] = ldfrag(Bt, K, col0 + t * 16, kn);
    __builtin_prefetch(A + (size_t)(row0 + (lane & 15)) * K + kn + 32, 0, 1);

#pragma unroll
    for (int t = 0; t < 8; ++t) {
      acc0[t] = wmma_bf16(a0, b[t], acc0[t]);
      acc1[t] = wmma_bf16(a1, b[t], acc1[t]);
    }
    a0 = na0; a1 = na1;
#pragma unroll
    for (int t = 0; t < 8; ++t) b[t] = nb[t];
  }

  const int rb0 = row0 + ((lane >> 4) << 3);
#pragma unroll
  for (int t = 0; t < 8; ++t) {
    int col = col0 + t * 16 + (lane & 15);
    float bv = bias ? bias[col] : 0.0f;
#pragma unroll
    for (int r = 0; r < 8; ++r) {
      float v0 = acc0[t][r] + bv;
      float v1 = acc1[t][r] + bv;
      if (act) { v0 = v0 > 0.0f ? v0 : 0.01f * v0; v1 = v1 > 0.0f ? v1 : 0.01f * v1; }
      size_t o0 = (size_t)(rb0 + r) * Nc + col;
      size_t o1 = (size_t)(rb0 + 16 + r) * Nc + col;
      if (Cf) { Cf[o0] = v0; Cf[o1] = v1; }
      if (Cb) { Cb[o0] = f2bf(v0); Cb[o1] = f2bf(v1); }
    }
  }
}

// ---------- split-K WMMA GEMM (for M=C_, K=N_): atomic f32 accumulate ----------
__global__ void __launch_bounds__(64) k_gemm_sk(const __bf16* __restrict__ A, const __bf16* __restrict__ Bt,
                                                float* __restrict__ Cf, int M, int Nc, int K, int kchunk) {
  const int lane = threadIdx.x & 31;
  const int wave = threadIdx.x >> 5;
  const int row0 = blockIdx.x * 32 + wave * 16;
  const int col0 = blockIdx.y * 128;
  const int k0 = blockIdx.z * kchunk, k1 = k0 + kchunk;
  v8f acc[8];
#pragma unroll
  for (int t = 0; t < 8; ++t) acc[t] = vzero8();

  v16bf a = ldfrag(A, K, row0, k0);
  v16bf b[8];
#pragma unroll
  for (int t = 0; t < 8; ++t) b[t] = ldfrag(Bt, K, col0 + t * 16, k0);

  for (int kc = k0; kc < k1; kc += 32) {
    int kn = (kc + 32 < k1) ? kc + 32 : kc;
    v16bf na = ldfrag(A, K, row0, kn);
    v16bf nb[8];
#pragma unroll
    for (int t = 0; t < 8; ++t) nb[t] = ldfrag(Bt, K, col0 + t * 16, kn);
    __builtin_prefetch(A + (size_t)(row0 + (lane & 15)) * K + kn + 32, 0, 1);

#pragma unroll
    for (int t = 0; t < 8; ++t) acc[t] = wmma_bf16(a, b[t], acc[t]);
    a = na;
#pragma unroll
    for (int t = 0; t < 8; ++t) b[t] = nb[t];
  }
  const int rb = row0 + ((lane >> 4) << 3);
#pragma unroll
  for (int t = 0; t < 8; ++t) {
    int col = col0 + t * 16 + (lane & 15);
#pragma unroll
    for (int r = 0; r < 8; ++r)
      atomicAdd(&Cf[(size_t)(rb + r) * Nc + col], acc[t][r]);
  }
}

// ---------- row statistics ----------
__global__ void __launch_bounds__(128) k_rowstats(const float* __restrict__ X, __bf16* __restrict__ nrm,
                                                  float* __restrict__ keepneg, float* __restrict__ flag) {
  int r = blockIdx.x, t = threadIdx.x;   // 128 threads, D = H_
  __shared__ float rs[128], rq[128];
  float v = X[(size_t)r * H_ + t];
  rs[t] = v; rq[t] = v * v;
  __syncthreads();
  for (int s = 64; s > 0; s >>= 1) {
    if (t < s) { rs[t] += rs[t + s]; rq[t] += rq[t + s]; }
    __syncthreads();
  }
  float nv = sqrtf(rq[0]);
  float sc = nv > 0.0f ? 1.0f / nv : 0.0f;
  if (nrm) nrm[(size_t)r * H_ + t] = f2bf(v * sc);
  if (t == 0) {
    if (keepneg) keepneg[r] = (rs[0] != 0.0f) ? 0.0f : -1e30f;
    if (flag)    flag[r]    = (nv > 0.0f) ? 1.0f : 0.0f;
  }
}

// ---------- softmax over axis 0 (stocks), [N,C] f32 -> transposed bf16 [C,N] ----------
__global__ void __launch_bounds__(256) k_softmax_col(const float* __restrict__ L, __bf16* __restrict__ outT) {
  int c = blockIdx.x, t = threadIdx.x;
  __shared__ float red[256];
  float m = -1e30f;
  for (int i = t; i < N_; i += 256) m = fmaxf(m, L[(size_t)i * C_ + c]);
  red[t] = m; __syncthreads();
  for (int s = 128; s > 0; s >>= 1) { if (t < s) red[t] = fmaxf(red[t], red[t + s]); __syncthreads(); }
  m = red[0]; __syncthreads();
  float sum = 0.0f;
  for (int i = t; i < N_; i += 256) sum += __expf(L[(size_t)i * C_ + c] - m);
  red[t] = sum; __syncthreads();
  for (int s = 128; s > 0; s >>= 1) { if (t < s) red[t] += red[t + s]; __syncthreads(); }
  float inv = 1.0f / red[0];
  for (int i = t; i < N_; i += 256)
    outT[(size_t)c * N_ + i] = f2bf(__expf(L[(size_t)i * C_ + c] - m) * inv);  // coalesced write
}

// ---------- masked row softmax over C (=512), f32 in -> bf16 out ----------
__global__ void __launch_bounds__(256) k_softmax_row(const float* __restrict__ L,
                                                     const float* __restrict__ keepneg,
                                                     __bf16* __restrict__ out) {
  int i = blockIdx.x, t = threadIdx.x;
  __shared__ float red[256];
  float v0 = L[(size_t)i * C_ + t]       + keepneg[t];
  float v1 = L[(size_t)i * C_ + t + 256] + keepneg[t + 256];
  red[t] = fmaxf(v0, v1); __syncthreads();
  for (int s = 128; s > 0; s >>= 1) { if (t < s) red[t] = fmaxf(red[t], red[t + s]); __syncthreads(); }
  float m = red[0]; __syncthreads();
  float e0 = v0 > -1e29f ? __expf(v0 - m) : 0.0f;
  float e1 = v1 > -1e29f ? __expf(v1 - m) : 0.0f;
  red[t] = e0 + e1; __syncthreads();
  for (int s = 128; s > 0; s >>= 1) { if (t < s) red[t] += red[t + s]; __syncthreads(); }
  float inv = red[0] > 0.0f ? 1.0f / red[0] : 0.0f;
  out[(size_t)i * C_ + t]       = f2bf(e0 * inv);
  out[(size_t)i * C_ + t + 256] = f2bf(e1 * inv);
}

// ---------- fused similarity + top-3 (diag forced to 0, [N,N] never materialized) ----------
__device__ __forceinline__ void ins3(float v, int j, float tv[3], int ti[3]) {
  if (v > tv[0])      { tv[2]=tv[1]; ti[2]=ti[1]; tv[1]=tv[0]; ti[1]=ti[0]; tv[0]=v; ti[0]=j; }
  else if (v > tv[1]) { tv[2]=tv[1]; ti[2]=ti[1]; tv[1]=v; ti[1]=j; }
  else if (v > tv[2]) { tv[2]=v; ti[2]=j; }
}

__global__ void __launch_bounds__(64) k_topk(const __bf16* __restrict__ hq,
                                             float* __restrict__ topv, int* __restrict__ topi) {
  const int lane = threadIdx.x & 31;
  const int wave = threadIdx.x >> 5;
  const int row0 = blockIdx.x * 64 + wave * 32;   // 32 rows per wave (2 tiles)
  v16bf a0[4], a1[4];
#pragma unroll
  for (int c = 0; c < 4; ++c) { a0[c] = ldfrag(hq, H_, row0, 32 * c); a1[c] = ldfrag(hq, H_, row0 + 16, 32 * c); }

  float tv[16][3]; int ti[16][3];
#pragma unroll
  for (int r = 0; r < 16; ++r)
#pragma unroll
    for (int k = 0; k < 3; ++k) { tv[r][k] = -1e30f; ti[r][k] = -1; }

  v16bf b[4];
#pragma unroll
  for (int c = 0; c < 4; ++c) b[c] = ldfrag(hq, H_, 0, 32 * c);

  const int rb0 = row0 + ((lane >> 4) << 3);
  for (int kb = 0; kb < N_; kb += 16) {
    int kn = (kb + 16 < N_) ? kb + 16 : kb;        // double-buffer next key tile
    v16bf nb[4];
#pragma unroll
    for (int c = 0; c < 4; ++c) nb[c] = ldfrag(hq, H_, kn, 32 * c);

    v8f s0 = vzero8(), s1 = vzero8();
#pragma unroll
    for (int c = 0; c < 4; ++c) {
      s0 = wmma_bf16(a0[c], b[c], s0);
      s1 = wmma_bf16(a1[c], b[c], s1);
    }
    int col = kb + (lane & 15);
#pragma unroll
    for (int r = 0; r < 8; ++r) {
      float v0 = s0[r]; if (col == rb0 + r)      v0 = 0.0f;   // eye mask
      float v1 = s1[r]; if (col == rb0 + 16 + r) v1 = 0.0f;
      ins3(v0, col, tv[r],     ti[r]);
      ins3(v1, col, tv[r + 8], ti[r + 8]);
    }
#pragma unroll
    for (int c = 0; c < 4; ++c) b[c] = nb[c];
  }
  for (int m = 1; m <= 8; m <<= 1) {
#pragma unroll
    for (int r = 0; r < 16; ++r) {
      float ov[3]; int oi[3];
#pragma unroll
      for (int k = 0; k < 3; ++k) { ov[k] = __shfl_xor(tv[r][k], m); oi[k] = __shfl_xor(ti[r][k], m); }
#pragma unroll
      for (int k = 0; k < 3; ++k) ins3(ov[k], oi[k], tv[r], ti[r]);
    }
  }
  if ((lane & 15) == 0) {
#pragma unroll
    for (int r = 0; r < 16; ++r) {
      int grow = rb0 + (r < 8 ? r : 8 + r);   // r<8 -> rb0+r ; r>=8 -> rb0+16+(r-8)
#pragma unroll
      for (int k = 0; k < 3; ++k) {
        topv[(size_t)grow * 3 + k] = tv[r][k];
        topi[(size_t)grow * 3 + k] = ti[r][k];
      }
    }
  }
}

// ---------- sparse scatter: hidden3 = hs2c_masked^T @ h_shared, colsumN ----------
__global__ void __launch_bounds__(128) k_scatter(const float* __restrict__ topv, const int* __restrict__ topi,
                                                 const float* __restrict__ hsh, float* __restrict__ h3,
                                                 float* __restrict__ colsumN) {
  int i = blockIdx.x, t = threadIdx.x;
  float x = hsh[(size_t)i * H_ + t];
#pragma unroll
  for (int k = 0; k < 3; ++k) {
    float v = topv[(size_t)i * 3 + k];
    int   j = topi[(size_t)i * 3 + k];
    if (j >= 0) {
      atomicAdd(&h3[(size_t)j * H_ + t], v * x);
      if (t == 0) atomicAdd(&colsumN[j], v);
    }
  }
}

__global__ void __launch_bounds__(128) k_diagadd(const float* __restrict__ colsumN, const float* __restrict__ flag,
                                                 const float* __restrict__ hsh, float* __restrict__ h3) {
  int i = blockIdx.x, t = threadIdx.x;
  if (colsumN[i] != 0.0f && flag[i] != 0.0f)
    h3[(size_t)i * H_ + t] += hsh[(size_t)i * H_ + t];
}

// ---------- flash-style fused attention helpers ----------
__device__ __forceinline__ void online_update(v8f& s0, v8f& s1, float* rm, float* rs, v8f* acc,
                                              __bf16* Pw, int lane) {
#pragma unroll
  for (int r = 0; r < 8; ++r) {
    float m = fmaxf(s0[r], s1[r]);
    for (int msk = 1; msk <= 8; msk <<= 1) m = fmaxf(m, __shfl_xor(m, msk));
    float nm = fmaxf(rm[r], m);
    float scale = __expf(rm[r] - nm);
    float p0 = s0[r] > -1e29f ? __expf(s0[r] - nm) : 0.0f;
    float p1 = s1[r] > -1e29f ? __expf(s1[r] - nm) : 0.0f;
    float ts = p0 + p1;
    for (int msk = 1; msk <= 8; msk <<= 1) ts += __shfl_xor(ts, msk);
    rs[r] = rs[r] * scale + ts;
    rm[r] = nm;
#pragma unroll
    for (int t = 0; t < 8; ++t) acc[t][r] *= scale;
    int prow = r + ((lane >> 4) << 3);
    Pw[prow * 32 +      (lane & 15)] = f2bf(p0);
    Pw[prow * 32 + 16 + (lane & 15)] = f2bf(p1);
  }
}

// out = softmax(cos(hq,h3n)+mask) @ h3v ; 32 rows/wave, 64 rows/block
__global__ void __launch_bounds__(64) k_attn(const __bf16* __restrict__ Q, const __bf16* __restrict__ Kn,
                                             const __bf16* __restrict__ V, const float* __restrict__ keepneg,
                                             __bf16* __restrict__ Ob) {
  __shared__ __bf16 Vt[128 * 48];        // V^T tile: rows 96B apart (32B aligned)
  __shared__ __bf16 P[2][2][16 * 32];    // [wave][row-tile]
  const int lane = threadIdx.x & 31;
  const int wave = threadIdx.x >> 5;
  const int row0 = blockIdx.x * 64 + wave * 32;

  v16bf a0[4], a1[4];
#pragma unroll
  for (int c = 0; c < 4; ++c) { a0[c] = ldfrag(Q, H_, row0, 32 * c); a1[c] = ldfrag(Q, H_, row0 + 16, 32 * c); }

  v8f acc0[8], acc1[8];
#pragma unroll
  for (int t = 0; t < 8; ++t) { acc0[t] = vzero8(); acc1[t] = vzero8(); }
  float rm0[8], rs0[8], rm1[8], rs1[8];
#pragma unroll
  for (int r = 0; r < 8; ++r) { rm0[r] = rm1[r] = -1e30f; rs0[r] = rs1[r] = 0.0f; }

  for (int kb = 0; kb < N_; kb += 32) {
    __syncthreads();                                  // prior Vt reads done
    for (int e = threadIdx.x; e < 32 * 16; e += 64) { // vectorized transposed staging
      int key = e >> 4, hg = (e & 15) * 8;
      v8bf row = *(const v8bf*)(V + (size_t)(kb + key) * H_ + hg);
#pragma unroll
      for (int q = 0; q < 8; ++q) Vt[(hg + q) * 48 + key] = row[q];
    }
    __syncthreads();

    // clause-grouped key fragment loads, then 16 score WMMAs
    v16bf kb0[4], kb1[4];
#pragma unroll
    for (int c = 0; c < 4; ++c) {
      kb0[c] = ldfrag(Kn, H_, kb,      32 * c);
      kb1[c] = ldfrag(Kn, H_, kb + 16, 32 * c);
    }
    v8f s00 = vzero8(), s01 = vzero8(), s10 = vzero8(), s11 = vzero8();
#pragma unroll
    for (int c = 0; c < 4; ++c) {
      s00 = wmma_bf16(a0[c], kb0[c], s00);
      s01 = wmma_bf16(a0[c], kb1[c], s01);
      s10 = wmma_bf16(a1[c], kb0[c], s10);
      s11 = wmma_bf16(a1[c], kb1[c], s11);
    }
    float kn0 = keepneg[kb +      (lane & 15)];
    float kn1 = keepneg[kb + 16 + (lane & 15)];
#pragma unroll
    for (int r = 0; r < 8; ++r) { s00[r] += kn0; s01[r] += kn1; s10[r] += kn0; s11[r] += kn1; }

    online_update(s00, s01, rm0, rs0, acc0, &P[wave][0][0], lane);
    online_update(s10, s11, rm1, rs1, acc1, &P[wave][1][0], lane);

    v16bf pa0 = *(const v16bf*)&P[wave][0][(lane & 15) * 32 + ((lane >> 4) << 4)];
    v16bf pa1 = *(const v16bf*)&P[wave][1][(lane & 15) * 32 + ((lane >> 4) << 4)];
#pragma unroll
    for (int t = 0; t < 8; ++t) {
      v16bf bv = *(const v16bf*)&Vt[(t * 16 + (lane & 15)) * 48 + ((lane >> 4) << 4)];
      acc0[t] = wmma_bf16(pa0, bv, acc0[t]);
      acc1[t] = wmma_bf16(pa1, bv, acc1[t]);
    }
  }

  const int rb0 = row0 + ((lane >> 4) << 3);
#pragma unroll
  for (int r = 0; r < 8; ++r) {
    float i0 = rs0[r] > 0.0f ? 1.0f / rs0[r] : 0.0f;
    float i1 = rs1[r] > 0.0f ? 1.0f / rs1[r] : 0.0f;
#pragma unroll
    for (int t = 0; t < 8; ++t) {
      Ob[(size_t)(rb0 + r)      * H_ + t * 16 + (lane & 15)] = f2bf(acc0[t][r] * i0);
      Ob[(size_t)(rb0 + 16 + r) * H_ + t * 16 + (lane & 15)] = f2bf(acc1[t][r] * i1);
    }
  }
}

// ---------- final: pred = (out_ps+out_hs+out_indi) @ W_out + b_out ----------
__global__ void __launch_bounds__(128) k_final(const float* __restrict__ ps, const float* __restrict__ hs,
                                               const float* __restrict__ ind, const float* __restrict__ w,
                                               const float* __restrict__ b, float* __restrict__ out) {
  int i = blockIdx.x, t = threadIdx.x;
  __shared__ float red[128];
  size_t o = (size_t)i * H_ + t;
  red[t] = (ps[o] + hs[o] + ind[o]) * w[t];
  __syncthreads();
  for (int s = 64; s > 0; s >>= 1) { if (t < s) red[t] += red[t + s]; __syncthreads(); }
  if (t == 0) out[i] = red[0] + b[0];
}

// ---------- host ----------
extern "C" void kernel_launch(void* const* d_in, const int* in_sizes, int n_in,
                              void* d_out, int out_size, void* d_ws, size_t ws_size,
                              hipStream_t stream) {
  const float* x    = (const float*)d_in[0];
  const float* mv   = (const float*)d_in[1];
  const int*   cm   = (const int*)d_in[2];
  const float* W_ps = (const float*)d_in[3],  *b_ps = (const float*)d_in[4];
  const float* W_hs = (const float*)d_in[5],  *b_hs = (const float*)d_in[6];
  const float* W_psf= (const float*)d_in[7],  *b_psf= (const float*)d_in[8];
  const float* W_hsf= (const float*)d_in[9],  *b_hsf= (const float*)d_in[10];
  const float* W_psb= (const float*)d_in[11], *b_psb= (const float*)d_in[12];
  const float* W_hsb= (const float*)d_in[13], *b_hsb= (const float*)d_in[14];
  const float* W_in = (const float*)d_in[15], *b_in = (const float*)d_in[16];
  const float* W_out= (const float*)d_in[23], *b_out= (const float*)d_in[24];
  float* pred = (float*)d_out;

  uintptr_t base = (uintptr_t)d_ws; size_t off = 0;
  auto alloc = [&](size_t bytes) -> void* {
    off = (off + 255) & ~(size_t)255;
    void* p = (void*)(base + off); off += bytes; return p;
  };
  __bf16* xb      = (__bf16*)alloc((size_t)N_ * H_ * 2);
  __bf16* xT      = (__bf16*)alloc((size_t)H_ * N_ * 2);
  __bf16* xnb     = (__bf16*)alloc((size_t)N_ * H_ * 2);
  __bf16* bufCN   = (__bf16*)alloc((size_t)C_ * N_ * 2);   // s2cT -> s2c2T -> c2s
  float*  logits  = (float*)alloc((size_t)N_ * C_ * 4);
  float*  colsumC = (float*)alloc((size_t)C_ * 4);
  float*  hidden1 = (float*)alloc((size_t)C_ * H_ * 4);
  __bf16* hidden1b= (__bf16*)alloc((size_t)C_ * H_ * 2);
  float*  keep1   = (float*)alloc((size_t)C_ * 4);
  float*  hidden2 = (float*)alloc((size_t)C_ * H_ * 4);
  __bf16* h2n     = (__bf16*)alloc((size_t)C_ * H_ * 2);
  __bf16* h2T     = (__bf16*)alloc((size_t)H_ * C_ * 2);
  __bf16* p1b     = (__bf16*)alloc((size_t)N_ * H_ * 2);
  __bf16* psb     = (__bf16*)alloc((size_t)N_ * H_ * 2);
  float*  p_back  = (float*)alloc((size_t)N_ * H_ * 4);
  float*  out_ps  = (float*)alloc((size_t)N_ * H_ * 4);
  float*  hsh     = (float*)alloc((size_t)N_ * H_ * 4);
  __bf16* hq      = (__bf16*)alloc((size_t)N_ * H_ * 2);
  float*  nflag   = (float*)alloc((size_t)N_ * 4);
  float*  topv    = (float*)alloc((size_t)N_ * 3 * 4);
  int*    topi    = (int*)alloc((size_t)N_ * 3 * 4);
  float*  colsumN = (float*)alloc((size_t)N_ * 4);
  float*  hidden3 = (float*)alloc((size_t)N_ * H_ * 4);
  float*  keep2   = (float*)alloc((size_t)N_ * 4);
  __bf16* h3n     = (__bf16*)alloc((size_t)N_ * H_ * 2);
  __bf16* h3v     = (__bf16*)alloc((size_t)N_ * H_ * 2);
  __bf16* hpb     = (__bf16*)alloc((size_t)N_ * H_ * 2);
  __bf16* hib     = (__bf16*)alloc((size_t)N_ * H_ * 2);
  float*  h_back  = (float*)alloc((size_t)N_ * H_ * 4);
  float*  out_hs  = (float*)alloc((size_t)N_ * H_ * 4);
  __bf16* indivb  = (__bf16*)alloc((size_t)N_ * H_ * 2);
  float*  out_in  = (float*)alloc((size_t)N_ * H_ * 4);
  __bf16* WT[7];
  const float* Wsrc[7] = { W_ps, W_psb, W_psf, W_hs, W_hsb, W_hsf, W_in };
  for (int i = 0; i < 7; ++i) WT[i] = (__bf16*)alloc((size_t)H_ * H_ * 2);
  (void)ws_size; (void)n_in; (void)in_sizes; (void)out_size;

  auto gemm = [&](const __bf16* A, const __bf16* Bt, const float* bias, float* Cf, __bf16* Cb,
                  int M, int Nc, int K, int act) {
    dim3 g(M / 64, Nc / 128);
    k_gemm<<<g, 64, 0, stream>>>(A, Bt, bias, Cf, Cb, M, Nc, K, act);
  };
  auto gemm_sk = [&](const __bf16* A, const __bf16* Bt, float* Cf, int M, int Nc, int K, int S) {
    dim3 g(M / 32, Nc / 128, S);
    k_gemm_sk<<<g, 64, 0, stream>>>(A, Bt, Cf, M, Nc, K, K / S);
  };
  const int NH = N_ * H_, CN = C_ * N_, CH = C_ * H_;

  // conversions / transposes
  k_f2b<<<(NH + 255) / 256, 256, 0, stream>>>(x, xb, NH);
  k_tr_f2b<<<(NH + 255) / 256, 256, 0, stream>>>(x, xT, N_, H_);
  for (int i = 0; i < 7; ++i)
    k_tr_f2b<<<(H_ * H_ + 255) / 256, 256, 0, stream>>>(Wsrc[i], WT[i], H_, H_);

  // phase A: market-value weighted stock->concept
  k_zero<<<(C_ + 255) / 256, 256, 0, stream>>>(colsumC, C_);
  k_colsum<<<256, 256, 0, stream>>>(cm, mv, colsumC);
  k_s2cT<<<(CN + 255) / 256, 256, 0, stream>>>(cm, mv, colsumC, bufCN);
  k_zero<<<(CH + 255) / 256, 256, 0, stream>>>(hidden1, CH);
  gemm_sk(bufCN, xT, hidden1, C_, H_, N_, 16);
  k_f2b<<<(CH + 255) / 256, 256, 0, stream>>>(hidden1, hidden1b, CH);
  k_rowstats<<<C_, 128, 0, stream>>>(hidden1, nullptr, keep1, nullptr);

  // phase B: stock-softmax re-aggregation
  gemm(xb, hidden1b, nullptr, logits, nullptr, N_, C_, H_, 0);
  k_softmax_col<<<C_, 256, 0, stream>>>(logits, bufCN);            // fused transpose out
  k_zero<<<(CH + 255) / 256, 256, 0, stream>>>(hidden2, CH);
  gemm_sk(bufCN, xT, hidden2, C_, H_, N_, 16);
  k_rowstats<<<C_, 128, 0, stream>>>(hidden2, h2n, nullptr, nullptr);
  k_tr_f2b<<<(CH + 255) / 256, 256, 0, stream>>>(hidden2, h2T, C_, H_);

  // phase C: concept cos-sim -> masked softmax -> p_shared branch
  k_rowstats<<<N_, 128, 0, stream>>>(x, xnb, nullptr, nullptr);
  gemm(xnb, h2n, nullptr, logits, nullptr, N_, C_, H_, 0);
  k_softmax_row<<<N_, 256, 0, stream>>>(logits, keep1, bufCN);
  gemm(bufCN, h2T, nullptr, nullptr, p1b, N_, H_, C_, 0);
  gemm(p1b, WT[0], b_ps,  nullptr, psb,    N_, H_, H_, 0);
  gemm(psb, WT[1], b_psb, p_back,  nullptr, N_, H_, H_, 0);
  gemm(psb, WT[2], b_psf, out_ps,  nullptr, N_, H_, H_, 1);

  // phase D: hidden-similarity branch (fused, [N,N] never materialized)
  k_sub2<<<(NH + 255) / 256, 256, 0, stream>>>(x, p_back, hsh, NH);
  k_rowstats<<<N_, 128, 0, stream>>>(hsh, hq, nullptr, nflag);
  k_topk<<<N_ / 64, 64, 0, stream>>>(hq, topv, topi);
  k_zero<<<(N_ + 255) / 256, 256, 0, stream>>>(colsumN, N_);
  k_zero<<<(NH + 255) / 256, 256, 0, stream>>>(hidden3, NH);
  k_scatter<<<N_, 128, 0, stream>>>(topv, topi, hsh, hidden3, colsumN);
  k_diagadd<<<N_, 128, 0, stream>>>(colsumN, nflag, hsh, hidden3);
  k_rowstats<<<N_, 128, 0, stream>>>(hidden3, h3n, keep2, nullptr);
  k_f2b<<<(NH + 255) / 256, 256, 0, stream>>>(hidden3, h3v, NH);
  k_attn<<<N_ / 64, 64, 0, stream>>>(hq, h3n, h3v, keep2, hpb);
  gemm(hpb, WT[3], b_hs,  nullptr, hib,    N_, H_, H_, 0);
  gemm(hib, WT[4], b_hsb, h_back,  nullptr, N_, H_, H_, 0);
  gemm(hib, WT[5], b_hsf, out_hs,  nullptr, N_, H_, H_, 1);

  // phase E: individual branch + head
  k_indiv<<<(NH + 255) / 256, 256, 0, stream>>>(x, p_back, h_back, indivb, NH);
  gemm(indivb, WT[6], b_in, out_in, nullptr, N_, H_, H_, 1);
  k_final<<<N_, 128, 0, stream>>>(out_ps, out_hs, out_in, W_out, b_out, pred);
}
